// PairformerLayer_61263413510628
// MI455X (gfx1250) — matrix operand order repaired and checked by reading, not compiled, not run on hardware
//
#include <hip/hip_runtime.h>
#include <hip/hip_bf16.h>

// ---------------------------------------------------------------------------
// Pairformer layer (AF3-style) for MI455X / gfx1250.
// GEMM-shaped work -> v_wmma_f32_16x16x32_f16; LDS tiles filled with
// global_load_async_to_lds_b128 (ASYNCcnt) per CDNA5 async-memory path.
// ---------------------------------------------------------------------------

#define NTOK 256
#define DZC  128      // edge channel dim
#define SDIM 384      // node dim
#define HNH  16       // node heads
#define DNH  24       // node head dim
#define HPH  4        // pair heads
#define CPW  32       // pair head width
#define BIGNEG 1e9f

typedef __attribute__((ext_vector_type(16))) _Float16 v16h;
typedef __attribute__((ext_vector_type(8)))  float    v8f;

// ---------------------------------------------------------------------------
// WMMA fragment helpers (CDNA5 16x16x32 f16 layouts, wave32)
// A (16x32, MxK): lanes 0-15 -> M=lane, K=0..15 (contiguous order used here, k = 16h + s);
//                 lanes 16-31 -> K=16..31.
// B (32x16, KxN): lanes 0-15 -> N=lane, K=0..15; lanes 16-31 -> K=16..31.
// C/D (16x16):    VGPR r, lanes 0-15 -> M=r, N=lane; lanes 16-31 -> M=r+8.
// ---------------------------------------------------------------------------

__device__ __forceinline__ v8f wmma_f16(v16h a, v16h b, v8f c) {
  return __builtin_amdgcn_wmma_f32_16x16x32_f16(false, a, false, b, (short)0, c,
                                                false, false);
}

// A fragment from f32 storage (LDS or global), converting to f16 in the gather.
__device__ __forceinline__ v16h frag_a_f32(const float* A, int lda) {
  const int lane = threadIdx.x & 31;
  const int m  = lane & 15;
  const int kb = (lane < 16) ? 0 : 16;
  v16h f;
#pragma unroll
  for (int v = 0; v < 8; ++v) {
    const int k = kb + 2 * v;
    f[2 * v]     = (_Float16)A[m * lda + k];
    f[2 * v + 1] = (_Float16)A[m * lda + k + 1];
  }
  return f;
}

// B fragment from f32 storage.
__device__ __forceinline__ v16h frag_b_f32(const float* B, int ldb) {
  const int lane = threadIdx.x & 31;
  const int n  = lane & 15;
  const int kb = (lane < 16) ? 0 : 16;
  v16h f;
#pragma unroll
  for (int h = 0; h < 16; ++h) f[h] = (_Float16)B[(kb + h) * ldb + n];
  return f;
}

// CDNA5 async copy: 16B global -> LDS, tracked by ASYNCcnt (no VGPR round-trip).
__device__ __forceinline__ void async_copy_b128(const float* gsrc, float* lds_dst) {
  const unsigned dst = (unsigned)(uintptr_t)lds_dst;               // LDS byte offset
  const unsigned long long src = (unsigned long long)(uintptr_t)gsrc;
  asm volatile("global_load_async_to_lds_b128 %0, %1, off" :: "v"(dst), "v"(src)
               : "memory");
}
__device__ __forceinline__ void async_wait0() {
  asm volatile("s_wait_asynccnt 0x0" ::: "memory");
}

__device__ __forceinline__ float sigf(float x) { return 1.f / (1.f + __expf(-x)); }

// ---------------------------------------------------------------------------
// LayerNorm: one block per row. swapN!=0 writes transposed (row i*N+j -> j*N+i).
// ---------------------------------------------------------------------------
__global__ void ln_kernel(const float* __restrict__ x, const float* __restrict__ g,
                          const float* __restrict__ b, float* __restrict__ y,
                          int D, int swapN) {
  __shared__ float red[128];
  const int row = blockIdx.x;
  const float* xr = x + (size_t)row * D;
  float s = 0.f;
  for (int i = threadIdx.x; i < D; i += 128) s += xr[i];
  red[threadIdx.x] = s; __syncthreads();
  for (int off = 64; off > 0; off >>= 1) {
    if ((int)threadIdx.x < off) red[threadIdx.x] += red[threadIdx.x + off];
    __syncthreads();
  }
  const float mean = red[0] / D;
  __syncthreads();
  float var = 0.f;
  for (int i = threadIdx.x; i < D; i += 128) { float d = xr[i] - mean; var += d * d; }
  red[threadIdx.x] = var; __syncthreads();
  for (int off = 64; off > 0; off >>= 1) {
    if ((int)threadIdx.x < off) red[threadIdx.x] += red[threadIdx.x + off];
    __syncthreads();
  }
  const float rstd = rsqrtf(red[0] / D + 1e-5f);
  size_t orow = row;
  if (swapN) { int i = row / swapN, j = row % swapN; orow = (size_t)j * swapN + i; }
  float* yr = y + orow * D;
  for (int i = threadIdx.x; i < D; i += 128)
    yr[i] = (xr[i] - mean) * rstd * g[i] + b[i];
}

// ---------------------------------------------------------------------------
// Generic WMMA GEMM: Y(=|+=) act(X[M,K] @ W[K,N] + bias) [* gate]
// block = 128 threads = 4 waves; tile 64(M) x 64(N); K chunks of 32.
// Each wave computes a 16x64 strip: 4 WMMAs share one A fragment per chunk.
// LDS tiles filled with async b128 copies (raw f32), f16 conversion in gather.
// act: 0 none, 1 sigmoid. swapN!=0: output row i*N+j written to j*N+i.
// Requires M%64==0, N%64==0, K%32==0.
// ---------------------------------------------------------------------------
#define LDA_F 36   // 64x32 A tile pitch (floats); row pitch 144B (16B aligned)
#define LDB_F 68   // 32x64 B tile pitch (floats); row pitch 272B (16B aligned)

__global__ void gemm_kernel(const float* __restrict__ X, const float* __restrict__ W,
                            const float* __restrict__ bias, const float* __restrict__ gate,
                            float* __restrict__ Y, int M, int K, int Nn,
                            int act, int resid, int swapN) {
  __shared__ __attribute__((aligned(16))) float sA[64 * LDA_F];
  __shared__ __attribute__((aligned(16))) float sB[32 * LDB_F];
  const int wave = threadIdx.x >> 5;
  const int lane = threadIdx.x & 31;
  const int m0 = blockIdx.x * 64;
  const int n0 = blockIdx.y * 64;
  v8f acc[4] = {{}, {}, {}, {}};
  for (int kt = 0; kt < K; kt += 32) {
#pragma unroll
    for (int it = 0; it < 4; ++it) {          // A: 64x32 floats, 16B per thread
      const int e = (it * 128 + threadIdx.x) * 4;
      const int r = e >> 5, c = e & 31;
      async_copy_b128(&X[(size_t)(m0 + r) * K + kt + c], &sA[r * LDA_F + c]);
    }
#pragma unroll
    for (int it = 0; it < 4; ++it) {          // B: 32x64 floats
      const int e = (it * 128 + threadIdx.x) * 4;
      const int r = e >> 6, c = e & 63;
      async_copy_b128(&W[(size_t)(kt + r) * Nn + n0 + c], &sB[r * LDB_F + c]);
    }
    async_wait0();
    __syncthreads();
    const v16h a = frag_a_f32(&sA[wave * 16 * LDA_F], LDA_F);
#pragma unroll
    for (int nt = 0; nt < 4; ++nt) {
      const v16h b = frag_b_f32(&sB[nt * 16], LDB_F);
      acc[nt] = wmma_f16(a, b, acc[nt]);
    }
    __syncthreads();
  }
  const int mrow = m0 + wave * 16 + (lane >> 4) * 8;
#pragma unroll
  for (int nt = 0; nt < 4; ++nt) {
    const int col = n0 + nt * 16 + (lane & 15);
#pragma unroll
    for (int r = 0; r < 8; ++r) {
      const int row = mrow + r;
      float v = acc[nt][r];
      if (bias) v += bias[col];
      if (act == 1) v = sigf(v);
      if (gate) v *= gate[(size_t)row * Nn + col];
      size_t orow = row;
      if (swapN) { int i = row / swapN, j = row % swapN; orow = (size_t)j * swapN + i; }
      const size_t idx = orow * Nn + col;
      if (resid) Y[idx] += v; else Y[idx] = v;
    }
  }
}

// ---------------------------------------------------------------------------
// Fused dual GEMM: Y = f(X@W1 + b1) * (X@W2 + b2) [* rowscale[row]]
// act: 0 -> sigmoid(t1)*t2 (triangle-mul branches); 1 -> silu(t1)*t2 (SwiGLU).
// 8 WMMAs per K-chunk share one A fragment. Same tiling/async staging as above.
// ---------------------------------------------------------------------------
__global__ void dualgemm_kernel(const float* __restrict__ X,
                                const float* __restrict__ W1, const float* __restrict__ b1,
                                const float* __restrict__ W2, const float* __restrict__ b2,
                                const float* __restrict__ rowscale, float* __restrict__ Y,
                                int M, int K, int Nn, int act) {
  __shared__ __attribute__((aligned(16))) float sA[64 * LDA_F];
  __shared__ __attribute__((aligned(16))) float sB1[32 * LDB_F];
  __shared__ __attribute__((aligned(16))) float sB2[32 * LDB_F];
  const int wave = threadIdx.x >> 5;
  const int lane = threadIdx.x & 31;
  const int m0 = blockIdx.x * 64;
  const int n0 = blockIdx.y * 64;
  v8f acc1[4] = {{}, {}, {}, {}};
  v8f acc2[4] = {{}, {}, {}, {}};
  for (int kt = 0; kt < K; kt += 32) {
#pragma unroll
    for (int it = 0; it < 4; ++it) {
      const int e = (it * 128 + threadIdx.x) * 4;
      const int r = e >> 5, c = e & 31;
      async_copy_b128(&X[(size_t)(m0 + r) * K + kt + c], &sA[r * LDA_F + c]);
    }
#pragma unroll
    for (int it = 0; it < 4; ++it) {
      const int e = (it * 128 + threadIdx.x) * 4;
      const int r = e >> 6, c = e & 63;
      async_copy_b128(&W1[(size_t)(kt + r) * Nn + n0 + c], &sB1[r * LDB_F + c]);
      async_copy_b128(&W2[(size_t)(kt + r) * Nn + n0 + c], &sB2[r * LDB_F + c]);
    }
    async_wait0();
    __syncthreads();
    const v16h a = frag_a_f32(&sA[wave * 16 * LDA_F], LDA_F);
#pragma unroll
    for (int nt = 0; nt < 4; ++nt) {
      const v16h f1 = frag_b_f32(&sB1[nt * 16], LDB_F);
      const v16h f2 = frag_b_f32(&sB2[nt * 16], LDB_F);
      acc1[nt] = wmma_f16(a, f1, acc1[nt]);
      acc2[nt] = wmma_f16(a, f2, acc2[nt]);
    }
    __syncthreads();
  }
  const int mrow = m0 + wave * 16 + (lane >> 4) * 8;
#pragma unroll
  for (int nt = 0; nt < 4; ++nt) {
    const int col = n0 + nt * 16 + (lane & 15);
#pragma unroll
    for (int r = 0; r < 8; ++r) {
      const int row = mrow + r;
      float t1 = acc1[nt][r] + (b1 ? b1[col] : 0.f);
      float t2 = acc2[nt][r] + (b2 ? b2[col] : 0.f);
      float v = (act == 0) ? (sigf(t1) * t2) : (t1 * sigf(t1) * t2);
      if (rowscale) v *= rowscale[row];
      Y[(size_t)row * Nn + col] = v;
    }
  }
}

// ---------------------------------------------------------------------------
// Triangle einsum:
//  outgoing: o[i,j,c] = sum_k a[i,k,c] * b[j,k,c]
//  incoming: o[i,j,c] = sum_k a[k,i,c] * b[k,j,c]
// grid (N/16, N/16, C); one wave per 16x16 tile of one channel.
// ---------------------------------------------------------------------------
__global__ void tri_einsum_kernel(const float* __restrict__ A, const float* __restrict__ Bv,
                                  float* __restrict__ O, int incoming) {
  const int i0 = blockIdx.x * 16, j0 = blockIdx.y * 16, c = blockIdx.z;
  const int lane = threadIdx.x;
  v8f acc = {};
  for (int k0 = 0; k0 < NTOK; k0 += 32) {
    v16h fa, fb;
    {
      const int m  = lane & 15;
      const int kb = (lane < 16) ? 0 : 16;
#pragma unroll
      for (int v = 0; v < 8; ++v) {
        const int k = kb + 2 * v;
#pragma unroll
        for (int u = 0; u < 2; ++u) {
          const int kk = k0 + k + u;
          const size_t ia = incoming ? ((size_t)kk * NTOK + (i0 + m))
                                     : ((size_t)(i0 + m) * NTOK + kk);
          fa[2 * v + u] = (_Float16)A[ia * DZC + c];
        }
      }
    }
    {
      const int n  = lane & 15;
      const int kb = (lane < 16) ? 0 : 16;
#pragma unroll
      for (int h = 0; h < 16; ++h) {
        const int kk = k0 + kb + h;
        const size_t ib = incoming ? ((size_t)kk * NTOK + (j0 + n))
                                   : ((size_t)(j0 + n) * NTOK + kk);
        fb[h] = (_Float16)Bv[ib * DZC + c];
      }
    }
    acc = wmma_f16(fa, fb, acc);
  }
  const int mr  = (lane >> 4) * 8;
  const int col = j0 + (lane & 15);
#pragma unroll
  for (int r = 0; r < 8; ++r)
    O[((size_t)(i0 + mr + r) * NTOK + col) * DZC + c] = acc[r];
}

// ---------------------------------------------------------------------------
// Triangle attention (flash-style, WMMA). One wave per (i, j-tile, head).
//  scores[j,k] = (q.k)/sqrt(32) + tb[j,k,h] + (mask-1)*1e9; softmax over k;
//  og = gate * (P @ V).  tmask!=0 indexes mask transposed (ending-node).
// ---------------------------------------------------------------------------
__global__ void tri_attn_kernel(const float* __restrict__ q, const float* __restrict__ k,
                                const float* __restrict__ v, const float* __restrict__ tb,
                                const float* __restrict__ gate, const float* __restrict__ mask,
                                float* __restrict__ og, int tmask) {
  __shared__ float sc[16][NTOK];
  const int i  = blockIdx.x;
  const int j0 = blockIdx.y * 16;
  const int h  = blockIdx.z;
  const int lane = threadIdx.x;
  const float scale = 0.17677669529663687f;  // 1/sqrt(32)

  // Q fragment: A[m][d] = q[i, j0+m, h*32+d]  (loaded once; K = d = 32 exactly)
  v16h fa;
  {
    const int m  = lane & 15;
    const int kb = (lane < 16) ? 0 : 16;
    const float* qr = q + ((size_t)i * NTOK + j0 + m) * (HPH * CPW) + h * CPW;
#pragma unroll
    for (int vv = 0; vv < 8; ++vv) {
      const int d = kb + 2 * vv;
      fa[2 * vv]     = (_Float16)qr[d];
      fa[2 * vv + 1] = (_Float16)qr[d + 1];
    }
  }
  // Phase 1: scores (16 k-tiles, one WMMA each)
  for (int kt = 0; kt < NTOK; kt += 16) {
    v16h fb;
    const int n  = lane & 15;
    const int kb = (lane < 16) ? 0 : 16;
#pragma unroll
    for (int hh = 0; hh < 16; ++hh) {
      const int d = kb + hh;
      fb[hh] = (_Float16)k[((size_t)i * NTOK + kt + n) * (HPH * CPW) + h * CPW + d];
    }
    v8f accs = {};
    accs = wmma_f16(fa, fb, accs);
    const int mr  = (lane >> 4) * 8;
    const int col = kt + (lane & 15);
#pragma unroll
    for (int r = 0; r < 8; ++r) {
      const int j = j0 + mr + r;
      const float mval = tmask ? mask[(size_t)col * NTOK + i]
                               : mask[(size_t)i * NTOK + col];
      sc[mr + r][col] = accs[r] * scale +
                        tb[((size_t)j * NTOK + col) * HPH + h] +
                        (mval - 1.f) * BIGNEG;
    }
  }
  __syncthreads();
  // Phase 2: row softmax (lanes 0..15, one row each), normalized probs in sc
  if (lane < 16) {
    float* row = sc[lane];
    float mx = -3.0e38f;
    for (int c = 0; c < NTOK; ++c) mx = fmaxf(mx, row[c]);
    float sum = 0.f;
    for (int c = 0; c < NTOK; ++c) { const float e = __expf(row[c] - mx); row[c] = e; sum += e; }
    const float inv = 1.f / sum;
    for (int c = 0; c < NTOK; ++c) row[c] *= inv;
  }
  __syncthreads();
  // Phase 3: O = P @ V (two 16-wide d halves, 8 WMMAs each)
#pragma unroll
  for (int nh = 0; nh < 2; ++nh) {
    v8f acco = {};
    for (int kt = 0; kt < NTOK; kt += 32) {
      const v16h pa = frag_a_f32(&sc[0][kt], NTOK);
      v16h fb;
      const int n  = lane & 15;
      const int kb = (lane < 16) ? 0 : 16;
#pragma unroll
      for (int hh = 0; hh < 16; ++hh) {
        const int kk = kt + kb + hh;
        fb[hh] = (_Float16)v[((size_t)i * NTOK + kk) * (HPH * CPW) + h * CPW + nh * 16 + n];
      }
      acco = wmma_f16(pa, fb, acco);
    }
    const int mr   = (lane >> 4) * 8;
    const int dcol = h * CPW + nh * 16 + (lane & 15);
#pragma unroll
    for (int r = 0; r < 8; ++r) {
      const size_t idx = ((size_t)i * NTOK + j0 + mr + r) * (HPH * CPW) + dcol;
      og[idx] = gate[idx] * acco[r];
    }
  }
}

// ---------------------------------------------------------------------------
// Tiny projection: Y[M,Nn] = X[M,K] @ W[K,Nn]; one output element per thread.
// Used for Nn=4 (triangle bias) and Nn=16 (pair-bias heads).
// ---------------------------------------------------------------------------
__global__ void small_proj_kernel(const float* __restrict__ X, const float* __restrict__ W,
                                  float* __restrict__ Y, int K, int Nn) {
  const size_t idx = (size_t)blockIdx.x * blockDim.x + threadIdx.x;
  const size_t m = idx / Nn;
  const int n = (int)(idx % Nn);
  const float* xr = X + m * K;
  float s = 0.f;
  for (int kk = 0; kk < K; ++kk) s += xr[kk] * W[kk * Nn + n];
  Y[idx] = s;
}

// ---------------------------------------------------------------------------
// Node attention with pair bias (d=24 -> VALU). grid (q, head), block 256 = #keys.
// og = sigmoid-gate * softmax(q.k/sqrt(24) + zb + maskbias) @ v
// ---------------------------------------------------------------------------
__global__ void node_attn_kernel(const float* __restrict__ qn, const float* __restrict__ kn,
                                 const float* __restrict__ vn, const float* __restrict__ zb,
                                 const float* __restrict__ gn, const float* __restrict__ nmask,
                                 float* __restrict__ og) {
  __shared__ float red[256];
  const int qi = blockIdx.x, h = blockIdx.y;
  const int t = threadIdx.x;
  const float scale = 0.20412414523193154f;  // 1/sqrt(24)
  const float* qr = qn + (size_t)qi * SDIM + h * DNH;
  const float* kr = kn + (size_t)t  * SDIM + h * DNH;
  float s = 0.f;
  for (int d = 0; d < DNH; ++d) s += qr[d] * kr[d];
  s = s * scale + zb[((size_t)qi * NTOK + t) * HNH + h] + (nmask[t] - 1.f) * BIGNEG;
  red[t] = s; __syncthreads();
  for (int off = 128; off > 0; off >>= 1) {
    if (t < off) red[t] = fmaxf(red[t], red[t + off]);
    __syncthreads();
  }
  const float mx = red[0]; __syncthreads();
  const float e = __expf(s - mx);
  red[t] = e; __syncthreads();
  for (int off = 128; off > 0; off >>= 1) {
    if (t < off) red[t] += red[t + off];
    __syncthreads();
  }
  const float pl = e / red[0]; __syncthreads();
  const float* vr = vn + (size_t)t * SDIM + h * DNH;
  for (int d = 0; d < DNH; ++d) {
    red[t] = pl * vr[d]; __syncthreads();
    for (int off = 128; off > 0; off >>= 1) {
      if (t < off) red[t] += red[t + off];
      __syncthreads();
    }
    if (t == 0) {
      const size_t idx = (size_t)qi * SDIM + h * DNH + d;
      og[idx] = gn[idx] * red[0];
    }
    __syncthreads();
  }
}

__global__ void copy_kernel(const float* __restrict__ src, float* __restrict__ dst, size_t n) {
  const size_t i = (size_t)blockIdx.x * blockDim.x + threadIdx.x;
  if (i < n) dst[i] = src[i];
}

// ---------------------------------------------------------------------------
// Host orchestration
// ---------------------------------------------------------------------------
namespace {

struct TriMulP {
  const float *Wag, *Wap, *Wbg, *Wbp, *Wg, *Wz;
  const float *bag, *bap, *bbg, *bbp, *bg, *bz;
  const float *ln_b, *ln_g, *lno_b, *lno_g;
};
struct TriAttnP {
  const float *Wb, *Wgt, *Wk, *Wo, *Wq, *Wv;
  const float *bgt, *bo, *ln_b, *ln_g;
};

inline void gemm(hipStream_t st, const float* X, const float* W, const float* bias,
                 const float* gate, float* Y, int M, int K, int Nn,
                 int act, int resid, int swapN) {
  gemm_kernel<<<dim3(M / 64, Nn / 64), dim3(128), 0, st>>>(X, W, bias, gate, Y, M, K, Nn,
                                                           act, resid, swapN);
}
inline void dualgemm(hipStream_t st, const float* X, const float* W1, const float* b1,
                     const float* W2, const float* b2, const float* rs, float* Y,
                     int M, int K, int Nn, int act) {
  dualgemm_kernel<<<dim3(M / 64, Nn / 64), dim3(128), 0, st>>>(X, W1, b1, W2, b2, rs, Y,
                                                               M, K, Nn, act);
}
inline void layernorm(hipStream_t st, const float* x, const float* g, const float* b,
                      float* y, int rows, int D, int swapN) {
  ln_kernel<<<dim3(rows), dim3(128), 0, st>>>(x, g, b, y, D, swapN);
}

}  // namespace

extern "C" void kernel_launch(void* const* d_in, const int* in_sizes, int n_in,
                              void* d_out, int out_size, void* d_ws, size_t ws_size,
                              hipStream_t stream) {
  (void)in_sizes; (void)n_in; (void)out_size; (void)ws_size;
  const int NN = NTOK * NTOK;

  // --- inputs: top-level in setup_inputs() insertion order -----------------
  const float* node_embed = (const float*)d_in[0];
  const float* edge_embed = (const float*)d_in[1];
  const float* node_mask  = (const float*)d_in[2];
  const float* edge_mask  = (const float*)d_in[3];

  // --- params: jax pytree flatten (dict keys sorted at every level) --------
  int p = 4;
  // apb: Wg Wk Wo Wq Wv Wz bq z_ln_b z_ln_g
  const float* apb_Wg = (const float*)d_in[p++];
  const float* apb_Wk = (const float*)d_in[p++];
  const float* apb_Wo = (const float*)d_in[p++];
  const float* apb_Wq = (const float*)d_in[p++];
  const float* apb_Wv = (const float*)d_in[p++];
  const float* apb_Wz = (const float*)d_in[p++];
  const float* apb_bq = (const float*)d_in[p++];
  const float* apb_zlb = (const float*)d_in[p++];
  const float* apb_zlg = (const float*)d_in[p++];
  // et: W1 W2 W3 ln_b ln_g
  const float* et_W1 = (const float*)d_in[p++];
  const float* et_W2 = (const float*)d_in[p++];
  const float* et_W3 = (const float*)d_in[p++];
  const float* et_lb = (const float*)d_in[p++];
  const float* et_lg = (const float*)d_in[p++];
  // nt: W1 W2 W3 ln_b ln_g
  const float* nt_W1 = (const float*)d_in[p++];
  const float* nt_W2 = (const float*)d_in[p++];
  const float* nt_W3 = (const float*)d_in[p++];
  const float* nt_lb = (const float*)d_in[p++];
  const float* nt_lg = (const float*)d_in[p++];
  const float* post_b = (const float*)d_in[p++];
  const float* post_g = (const float*)d_in[p++];
  const float* pre_b  = (const float*)d_in[p++];
  const float* pre_g  = (const float*)d_in[p++];
  // tae / tas: Wb Wgt Wk Wo Wq Wv bgt bo ln_b ln_g
  TriAttnP tae, tas;
  tae.Wb  = (const float*)d_in[p++]; tae.Wgt = (const float*)d_in[p++];
  tae.Wk  = (const float*)d_in[p++]; tae.Wo  = (const float*)d_in[p++];
  tae.Wq  = (const float*)d_in[p++]; tae.Wv  = (const float*)d_in[p++];
  tae.bgt = (const float*)d_in[p++]; tae.bo  = (const float*)d_in[p++];
  tae.ln_b = (const float*)d_in[p++]; tae.ln_g = (const float*)d_in[p++];
  tas.Wb  = (const float*)d_in[p++]; tas.Wgt = (const float*)d_in[p++];
  tas.Wk  = (const float*)d_in[p++]; tas.Wo  = (const float*)d_in[p++];
  tas.Wq  = (const float*)d_in[p++]; tas.Wv  = (const float*)d_in[p++];
  tas.bgt = (const float*)d_in[p++]; tas.bo  = (const float*)d_in[p++];
  tas.ln_b = (const float*)d_in[p++]; tas.ln_g = (const float*)d_in[p++];
  // tmi / tmo: Wag Wap Wbg Wbp Wg Wz bag bap bbg bbp bg bz ln_b ln_g lno_b lno_g
  TriMulP tmi, tmo;
  tmi.Wag = (const float*)d_in[p++]; tmi.Wap = (const float*)d_in[p++];
  tmi.Wbg = (const float*)d_in[p++]; tmi.Wbp = (const float*)d_in[p++];
  tmi.Wg  = (const float*)d_in[p++]; tmi.Wz  = (const float*)d_in[p++];
  tmi.bag = (const float*)d_in[p++]; tmi.bap = (const float*)d_in[p++];
  tmi.bbg = (const float*)d_in[p++]; tmi.bbp = (const float*)d_in[p++];
  tmi.bg  = (const float*)d_in[p++]; tmi.bz  = (const float*)d_in[p++];
  tmi.ln_b = (const float*)d_in[p++]; tmi.ln_g = (const float*)d_in[p++];
  tmi.lno_b = (const float*)d_in[p++]; tmi.lno_g = (const float*)d_in[p++];
  tmo.Wag = (const float*)d_in[p++]; tmo.Wap = (const float*)d_in[p++];
  tmo.Wbg = (const float*)d_in[p++]; tmo.Wbp = (const float*)d_in[p++];
  tmo.Wg  = (const float*)d_in[p++]; tmo.Wz  = (const float*)d_in[p++];
  tmo.bag = (const float*)d_in[p++]; tmo.bap = (const float*)d_in[p++];
  tmo.bbg = (const float*)d_in[p++]; tmo.bbp = (const float*)d_in[p++];
  tmo.bg  = (const float*)d_in[p++]; tmo.bz  = (const float*)d_in[p++];
  tmo.ln_b = (const float*)d_in[p++]; tmo.ln_g = (const float*)d_in[p++];
  tmo.lno_b = (const float*)d_in[p++]; tmo.lno_g = (const float*)d_in[p++];

  // --- outputs / workspace -------------------------------------------------
  float* outn = (float*)d_out;                             // [256,384]
  float* oute = (float*)d_out + (size_t)NTOK * SDIM;       // [256,256,128]
  float* ws = (float*)d_ws;
  const size_t EC = (size_t)NN * DZC;                      // 8.39M floats
  float* bufLN = ws;
  float* bufA  = ws + 1 * EC;
  float* bufB  = ws + 2 * EC;
  float* bufC  = ws + 3 * EC;
  float* bufD  = ws + 4 * EC;
  float* bufE  = ws + 5 * EC;
  float* hidden = bufA;                                    // spans bufA..bufD (4*EC)
  float* sm = ws + 6 * EC;
  float* tbbuf = sm;  sm += (size_t)NN * HPH;
  float* zbbuf = sm;  sm += (size_t)NN * HNH;
  float* snbuf = sm;  sm += (size_t)NTOK * SDIM;
  float* qnbuf = sm;  sm += (size_t)NTOK * SDIM;
  float* knbuf = sm;  sm += (size_t)NTOK * SDIM;
  float* vnbuf = sm;  sm += (size_t)NTOK * SDIM;
  float* gnbuf = sm;  sm += (size_t)NTOK * SDIM;
  float* onbuf = sm;  sm += (size_t)NTOK * SDIM;
  float* nbuf  = sm;  sm += (size_t)NTOK * SDIM;
  float* nhid  = sm;  sm += (size_t)NTOK * 4 * SDIM;

  // e := edge_embed (all triangle updates accumulate in place in d_out)
  copy_kernel<<<dim3((unsigned)(EC / 256)), dim3(256), 0, stream>>>(edge_embed, oute, EC);

  // ---- triangle multiplicative updates ------------------------------------
  auto run_tri_mul = [&](const TriMulP& P, int incoming) {
    layernorm(stream, oute, P.ln_g, P.ln_b, bufLN, NN, DZC, 0);
    dualgemm(stream, bufLN, P.Wag, P.bag, P.Wap, P.bap, edge_mask, bufA, NN, DZC, DZC, 0);
    dualgemm(stream, bufLN, P.Wbg, P.bbg, P.Wbp, P.bbp, edge_mask, bufB, NN, DZC, DZC, 0);
    tri_einsum_kernel<<<dim3(NTOK / 16, NTOK / 16, DZC), dim3(32), 0, stream>>>(
        bufA, bufB, bufC, incoming);
    layernorm(stream, bufC, P.lno_g, P.lno_b, bufD, NN, DZC, 0);
    gemm(stream, bufLN, P.Wg, P.bg, nullptr, bufE, NN, DZC, DZC, /*act=*/1, 0, 0);
    gemm(stream, bufD, P.Wz, P.bz, /*gate=*/bufE, oute, NN, DZC, DZC, 0, /*resid=*/1, 0);
  };
  run_tri_mul(tmo, /*incoming=*/0);
  run_tri_mul(tmi, /*incoming=*/1);

  // ---- triangle attention --------------------------------------------------
  auto run_tri_attn = [&](const TriAttnP& P, int starting) {
    const int swap = starting ? 0 : NTOK;
    layernorm(stream, oute, P.ln_g, P.ln_b, bufLN, NN, DZC, swap);  // swap -> T space
    gemm(stream, bufLN, P.Wq, nullptr, nullptr, bufA, NN, DZC, HPH * CPW, 0, 0, 0);
    gemm(stream, bufLN, P.Wk, nullptr, nullptr, bufB, NN, DZC, HPH * CPW, 0, 0, 0);
    gemm(stream, bufLN, P.Wv, nullptr, nullptr, bufC, NN, DZC, HPH * CPW, 0, 0, 0);
    gemm(stream, bufLN, P.Wgt, P.bgt, nullptr, bufD, NN, DZC, HPH * CPW, /*act=*/1, 0, 0);
    small_proj_kernel<<<dim3((unsigned)((size_t)NN * HPH / 256)), dim3(256), 0, stream>>>(
        bufLN, P.Wb, tbbuf, DZC, HPH);
    tri_attn_kernel<<<dim3(NTOK, NTOK / 16, HPH), dim3(32), 0, stream>>>(
        bufA, bufB, bufC, tbbuf, bufD, edge_mask, bufE, /*tmask=*/starting ? 0 : 1);
    gemm(stream, bufE, P.Wo, P.bo, nullptr, oute, NN, HPH * CPW, DZC, 0, /*resid=*/1, swap);
  };
  run_tri_attn(tas, 1);
  run_tri_attn(tae, 0);

  // ---- edge transition (SwiGLU) -------------------------------------------
  layernorm(stream, oute, et_lg, et_lb, bufLN, NN, DZC, 0);
  dualgemm(stream, bufLN, et_W1, nullptr, et_W2, nullptr, nullptr, hidden,
           NN, DZC, 4 * DZC, /*act=silu*/ 1);
  gemm(stream, hidden, et_W3, nullptr, nullptr, oute, NN, 4 * DZC, DZC, 0, /*resid=*/1, 0);

  // ---- node pathway: attention with pair bias -----------------------------
  layernorm(stream, node_embed, pre_g, pre_b, snbuf, NTOK, SDIM, 0);
  gemm(stream, snbuf, apb_Wq, apb_bq, nullptr, qnbuf, NTOK, SDIM, SDIM, 0, 0, 0);
  gemm(stream, snbuf, apb_Wk, nullptr, nullptr, knbuf, NTOK, SDIM, SDIM, 0, 0, 0);
  gemm(stream, snbuf, apb_Wv, nullptr, nullptr, vnbuf, NTOK, SDIM, SDIM, 0, 0, 0);
  gemm(stream, snbuf, apb_Wg, nullptr, nullptr, gnbuf, NTOK, SDIM, SDIM, /*act=*/1, 0, 0);
  layernorm(stream, oute, apb_zlg, apb_zlb, bufLN, NN, DZC, 0);
  small_proj_kernel<<<dim3((unsigned)((size_t)NN * HNH / 256)), dim3(256), 0, stream>>>(
      bufLN, apb_Wz, zbbuf, DZC, HNH);
  copy_kernel<<<dim3((unsigned)((size_t)NTOK * SDIM / 256)), dim3(256), 0, stream>>>(
      node_embed, nbuf, (size_t)NTOK * SDIM);
  node_attn_kernel<<<dim3(NTOK, HNH), dim3(256), 0, stream>>>(
      qnbuf, knbuf, vnbuf, zbbuf, gnbuf, node_mask, onbuf);
  gemm(stream, onbuf, apb_Wo, nullptr, nullptr, nbuf, NTOK, SDIM, SDIM, 0, /*resid=*/1, 0);

  // ---- node transition (SwiGLU) + post-LN ---------------------------------
  layernorm(stream, nbuf, nt_lg, nt_lb, snbuf, NTOK, SDIM, 0);
  dualgemm(stream, snbuf, nt_W1, nullptr, nt_W2, nullptr, nullptr, nhid,
           NTOK, SDIM, 4 * SDIM, /*act=silu*/ 1);
  gemm(stream, nhid, nt_W3, nullptr, nullptr, nbuf, NTOK, 4 * SDIM, SDIM, 0, /*resid=*/1, 0);
  layernorm(stream, nbuf, post_g, post_b, outn, NTOK, SDIM, 0);
}
